// DQN_10668698763736
// MI455X (gfx1250) — matrix operand order repaired and checked by reading, not compiled, not run on hardware
//
#include <hip/hip_runtime.h>
#include <math.h>

// GAT 4-layer forward for MI455X (gfx1250).
// - Dense transforms: V_WMMA_F32_16X16X32_F16; one block = 8 row-tiles x one
//   16-col band; the 16 x IN_C weight slab is staged into LDS once per block,
//   via GLOBAL_LOAD_ASYNC_TO_LDS_B128 (ASYNCcnt) when available.
// - Edge softmax/aggregation: L2-resident f32 atomic scatter (h/agg buffers
//   fit in the 192MB L2), b128 gathers.

typedef __attribute__((ext_vector_type(16))) _Float16 v16h;
typedef __attribute__((ext_vector_type(8)))  float    v8f;

#define HEADS 4
#define NNODE 20000
#define NEDGE 320000
#define ETOT  (NEDGE + NNODE)

#if defined(__gfx1250__) && __has_builtin(__builtin_amdgcn_global_load_async_to_lds_b128)
#define HAVE_ASYNC_LDS 1
// async-to-LDS builtin operand types (clang prints AS1 as "__device__"):
//   (v4i addrspace(1)*, v4i addrspace(3)*, imm offset, imm cpol)
typedef int v4i __attribute__((vector_size(16)));
typedef v4i __attribute__((address_space(1))) gv4i;
typedef v4i __attribute__((address_space(3))) lv4i;
#else
#define HAVE_ASYNC_LDS 0
#endif

// ---------- helpers ----------

__device__ __forceinline__ unsigned f32_ord(float f) {
  unsigned u = __float_as_uint(f);
  return (u & 0x80000000u) ? ~u : (u | 0x80000000u);
}
__device__ __forceinline__ float ord_f32(unsigned o) {
  unsigned u = (o & 0x80000000u) ? (o & 0x7fffffffu) : ~o;
  return __uint_as_float(u);
}
__device__ __forceinline__ unsigned hash_u32(unsigned x) {
  x ^= x >> 16; x *= 0x7feb352du;
  x ^= x >> 15; x *= 0x846ca68bu;
  x ^= x >> 16; return x;
}

// ---------- generic fill ----------

__global__ void fill_u32_kernel(unsigned* __restrict__ p, unsigned v, int count) {
  int i = blockIdx.x * blockDim.x + threadIdx.x;
  if (i < count) p[i] = v;
}

// ---------- WMMA GEMM: H[N, HC] = X[N, IN_C] @ W[HC, IN_C]^T ----------
// Block = 256 threads = 8 waves; each wave one 16x16 tile, all waves share a
// column band. The 16 x IN_C f32 weight slab lives in LDS (col-major), staged
// asynchronously; B fragments become contiguous ds_load_b128 per lane.

template <int IN_C, int HC>
__global__ __launch_bounds__(256) void gemm_wmma_kernel(
    const float* __restrict__ X, const float* __restrict__ W,
    float* __restrict__ H) {
  constexpr int NTC   = (HC + 15) / 16;
  constexpr int NROWT = NNODE / 16;          // 1250 row tiles (exact)
  __shared__ float wlds[16 * IN_C];          // wlds[col*IN_C + k]

  int colTile = blockIdx.x % NTC;
  int rowBlk  = blockIdx.x / NTC;
  int colBase = colTile * 16;
  int tid = threadIdx.x;

  // ---- stage weight slab: W[colBase+col][k], col 0..15, k 0..IN_C-1 ----
  constexpr int NV4 = (16 * IN_C) / 4;       // float4 chunks
  for (int i4 = tid; i4 < NV4; i4 += 256) {
    int col = i4 / (IN_C / 4);
    int kq  = i4 - col * (IN_C / 4);
    int gcol = colBase + col;
    gcol = (gcol < HC) ? gcol : (HC - 1);    // clamp; masked at fragment read
    const float* gsrc = W + (size_t)gcol * IN_C + kq * 4;
    float*       ldst = wlds + col * IN_C + kq * 4;
#if HAVE_ASYNC_LDS
    __builtin_amdgcn_global_load_async_to_lds_b128(
        (gv4i*)gsrc, (lv4i*)ldst, 0, 0);
#else
    *(float4*)ldst = *(const float4*)gsrc;
#endif
  }
#if HAVE_ASYNC_LDS
#if __has_builtin(__builtin_amdgcn_s_wait_asynccnt)
  __builtin_amdgcn_s_wait_asynccnt(0);
#else
  asm volatile("s_wait_asynccnt 0x0" ::: "memory");
#endif
#endif
  __syncthreads();                            // all waves staged before compute

  int waveId = tid >> 5;
  int rt = rowBlk * 8 + waveId;
  if (rt >= NROWT) return;                    // tail waves exit AFTER barrier

  int lane = tid & 31;
  int lo  = lane & 15;
  int hi  = lane >> 4;
  int hiA = hi * 8;                           // A K offset for high half-wave
  int hiB = hi * 16;                          // unused; B uses hiA, same K order as A
  int rowBase = rt * 16;

  const float* xr = X + (size_t)(rowBase + lo) * IN_C;
  bool colok = (colBase + lo) < HC;
  const float* bp = wlds + lo * IN_C;         // lane's B column (contiguous K)

  v8f acc = {};
#pragma unroll
  for (int k0 = 0; k0 < IN_C; k0 += 32) {
    float bufA[16], bufB[16];
    // A 16x32 f16 layout: low half-wave K {0..7,16..23}, high {8..15,24..31}.
#pragma unroll
    for (int r = 0; r < 2; r++) {
#pragma unroll
      for (int q = 0; q < 2; q++) {
        int  k  = k0 + hiA + r * 16 + q * 4;
        bool ok = (k + 4 <= IN_C);
        const float4 v = *(const float4*)(xr + (ok ? k : 0));
        int s = r * 8 + q * 4;
        bufA[s + 0] = ok ? v.x : 0.0f;
        bufA[s + 1] = ok ? v.y : 0.0f;
        bufA[s + 2] = ok ? v.z : 0.0f;
        bufA[s + 3] = ok ? v.w : 0.0f;
      }
    }
    // B 32x16 f16 layout: low half-wave K {0..7,16..23}, high {8..15,24..31}.
#pragma unroll
    for (int q = 0; q < 4; q++) {
      int  k   = k0 + hiA + (q >> 1) * 16 + (q & 1) * 4;
      bool kok = (k + 4 <= IN_C);
      const float4 v = *(const float4*)(bp + (kok ? k : 0));   // ds_load_b128
      bool ok = kok && colok;
      int s = q * 4;
      bufB[s + 0] = ok ? v.x : 0.0f;
      bufB[s + 1] = ok ? v.y : 0.0f;
      bufB[s + 2] = ok ? v.z : 0.0f;
      bufB[s + 3] = ok ? v.w : 0.0f;
    }
    v16h a, b;
#pragma unroll
    for (int i = 0; i < 16; i++) {
      a[i] = (_Float16)bufA[i];
      b[i] = (_Float16)bufB[i];
    }
    acc = __builtin_amdgcn_wmma_f32_16x16x32_f16(
        /*neg_a=*/false, a, /*neg_b=*/false, b,
        /*c_mod=*/(short)0, acc, /*reuse_a=*/false, /*reuse_b=*/false);
  }
  // D layout: VGPR r -> M = r (lanes 0-15) / r+8 (lanes 16-31), N = lane&15
  int srow = rowBase + hi * 8;
  int scol = colBase + lo;
  if (scol < HC) {
#pragma unroll
    for (int r = 0; r < 8; r++)
      H[(size_t)(srow + r) * HC + scol] = acc[r];
  }
}

// ---------- per-(node,head) attention logits ----------

template <int OUT_C>
__global__ void alpha_kernel(const float* __restrict__ h,
                             const float* __restrict__ a_src,
                             const float* __restrict__ a_dst,
                             float* __restrict__ as_out,
                             float* __restrict__ ad_out) {
  int idx = blockIdx.x * blockDim.x + threadIdx.x;
  if (idx >= NNODE * HEADS) return;
  int nd = idx / HEADS, hh = idx - nd * HEADS;
  const float* row = h + (size_t)nd * (HEADS * OUT_C) + hh * OUT_C;
  const float* vs = a_src + hh * OUT_C;
  const float* vd = a_dst + hh * OUT_C;
  float s = 0.0f, d = 0.0f;
  if (OUT_C % 4 == 0) {
#pragma unroll
    for (int c = 0; c < OUT_C; c += 4) {
      const float4 hv = *(const float4*)(row + c);
      const float4 sv = *(const float4*)(vs + c);
      const float4 dv = *(const float4*)(vd + c);
      s += hv.x * sv.x + hv.y * sv.y + hv.z * sv.z + hv.w * sv.w;
      d += hv.x * dv.x + hv.y * dv.y + hv.z * dv.z + hv.w * dv.w;
    }
  } else {
#pragma unroll
    for (int c = 0; c < OUT_C; c++) { float v = row[c]; s += v * vs[c]; d += v * vd[c]; }
  }
  as_out[idx] = s;
  ad_out[idx] = d;
}

// ---------- edge pass 1: e = leaky_relu(as[src]+ad[dst]); segment max ----------

__global__ void edge_max_kernel(const int* __restrict__ ei,
                                const float* __restrict__ as,
                                const float* __restrict__ ad,
                                float* __restrict__ eedge,
                                unsigned* __restrict__ emax) {
  int idx = blockIdx.x * blockDim.x + threadIdx.x;
  if (idx >= ETOT * HEADS) return;
  int e = idx / HEADS, hh = idx - e * HEADS;
  int src = (e < NEDGE) ? ei[e]         : (e - NEDGE);   // self-loop append
  int dst = (e < NEDGE) ? ei[NEDGE + e] : (e - NEDGE);
  float v = as[src * HEADS + hh] + ad[dst * HEADS + hh];
  v = (v > 0.0f) ? v : 0.2f * v;                         // leaky_relu 0.2
  eedge[idx] = v;
  atomicMax(emax + dst * HEADS + hh, f32_ord(v));
}

// ---------- edge pass 2: ee = exp(e - max); segment sum ----------

__global__ void edge_exp_kernel(const int* __restrict__ ei,
                                float* __restrict__ eedge,
                                const unsigned* __restrict__ emax,
                                float* __restrict__ denom) {
  int idx = blockIdx.x * blockDim.x + threadIdx.x;
  if (idx >= ETOT * HEADS) return;
  int e = idx / HEADS, hh = idx - e * HEADS;
  int dst = (e < NEDGE) ? ei[NEDGE + e] : (e - NEDGE);
  float m = ord_f32(emax[dst * HEADS + hh]);
  float ee = __expf(eedge[idx] - m);
  eedge[idx] = ee;
  atomicAdd(denom + dst * HEADS + hh, ee);
}

// ---------- edge pass 3: agg[dst] += h[src] * alpha ----------

template <int OUT_C>
__global__ void edge_scatter_kernel(const int* __restrict__ ei,
                                    const float* __restrict__ eedge,
                                    const float* __restrict__ denom,
                                    const float* __restrict__ h,
                                    float* __restrict__ agg) {
  constexpr int HC = HEADS * OUT_C;
  int idx = blockIdx.x * blockDim.x + threadIdx.x;
  if (idx >= ETOT * HEADS) return;
  int e = idx / HEADS, hh = idx - e * HEADS;
  int src = (e < NEDGE) ? ei[e]         : (e - NEDGE);
  int dst = (e < NEDGE) ? ei[NEDGE + e] : (e - NEDGE);
  float al = eedge[idx] / (denom[dst * HEADS + hh] + 1e-16f);
  const float* hrow = h   + (size_t)src * HC + hh * OUT_C;
  float*       arow = agg + (size_t)dst * HC + hh * OUT_C;
  __builtin_prefetch(hrow, 0, 0);   // global_prefetch_b8
  if (OUT_C % 4 == 0) {
#pragma unroll 4
    for (int c = 0; c < OUT_C; c += 4) {
      const float4 v = *(const float4*)(hrow + c);   // b128 gather (L2-resident)
      atomicAdd(arow + c + 0, v.x * al);
      atomicAdd(arow + c + 1, v.y * al);
      atomicAdd(arow + c + 2, v.z * al);
      atomicAdd(arow + c + 3, v.w * al);
    }
  } else {
#pragma unroll
    for (int c = 0; c < OUT_C; c++)
      atomicAdd(arow + c, hrow[c] * al);
  }
}

// ---------- head-mean + bias (+ batchnorm stats) ----------

__global__ void node_finalize_kernel(const float* __restrict__ agg,
                                     const float* __restrict__ bias,
                                     float* __restrict__ y,
                                     float* __restrict__ bnsum,
                                     float* __restrict__ bnsqs,
                                     int out_c, int hc, int do_stats) {
  int idx = blockIdx.x * blockDim.x + threadIdx.x;
  if (idx >= NNODE * out_c) return;
  int nd = idx / out_c, c = idx - nd * out_c;
  float s = 0.0f;
#pragma unroll
  for (int hh = 0; hh < HEADS; hh++)
    s += agg[(size_t)nd * hc + hh * out_c + c];
  float v = s * 0.25f + bias[c];
  y[idx] = v;
  if (do_stats) {
    atomicAdd(bnsum + c, v);
    atomicAdd(bnsqs + c, v * v);
  }
}

// ---------- batchnorm + ELU (+ dropout) ----------

__global__ void bn_act_kernel(const float* __restrict__ y,
                              const float* __restrict__ bnsum,
                              const float* __restrict__ bnsqs,
                              const float* __restrict__ gamma,
                              const float* __restrict__ beta,
                              float* __restrict__ xout,
                              int out_c, int do_drop, unsigned seed) {
  int idx = blockIdx.x * blockDim.x + threadIdx.x;
  if (idx >= NNODE * out_c) return;
  int c = idx % out_c;
  float mu  = bnsum[c] * (1.0f / NNODE);
  float var = bnsqs[c] * (1.0f / NNODE) - mu * mu;      // biased var
  float xn = (y[idx] - mu) * rsqrtf(var + 1e-5f) * gamma[c] + beta[c];
  xn = (xn > 0.0f) ? xn : (__expf(xn) - 1.0f);          // ELU
  if (do_drop) {
    unsigned r = hash_u32((unsigned)idx * 0x9E3779B9u + seed);
    xn = (r & 0x10000u) ? xn * 2.0f : 0.0f;             // p=0.5, scale 2
  }
  xout[idx] = xn;
}

// ---------- tanh + global softmax over N ----------

__global__ void tanh_max_kernel(const float* __restrict__ actor,
                                float* __restrict__ out,
                                unsigned* __restrict__ smax) {
  int i = blockIdx.x * blockDim.x + threadIdx.x;
  if (i >= NNODE) return;
  float l = tanhf(actor[i]);
  out[i] = l;
  atomicMax(smax, f32_ord(l));
}

__global__ void sum_exp_kernel(const float* __restrict__ out,
                               const unsigned* __restrict__ smax,
                               float* __restrict__ ssum) {
  int i = blockIdx.x * blockDim.x + threadIdx.x;
  if (i >= NNODE) return;
  float m = ord_f32(*smax);
  atomicAdd(ssum, __expf(out[i] - m));
}

__global__ void prob_kernel(float* __restrict__ out,
                            const unsigned* __restrict__ smax,
                            const float* __restrict__ ssum) {
  int i = blockIdx.x * blockDim.x + threadIdx.x;
  if (i >= NNODE) return;
  float m = ord_f32(*smax);
  out[NNODE + i] = __expf(out[i] - m) / (*ssum);
}

// ---------- host launcher ----------

static inline dim3 grid1d(long n) { return dim3((unsigned)((n + 255) / 256)); }

extern "C" void kernel_launch(void* const* d_in, const int* in_sizes, int n_in,
                              void* d_out, int out_size, void* d_ws, size_t ws_size,
                              hipStream_t stream) {
  (void)in_sizes; (void)n_in; (void)out_size; (void)ws_size;

  const float* x  = (const float*)d_in[0];
  const int*   ei = (const int*)d_in[1];   // [2,E] flat: src then dst (int32)

  const float* W[4]  = {(const float*)d_in[2],  (const float*)d_in[8],
                        (const float*)d_in[14], (const float*)d_in[20]};
  const float* AS[4] = {(const float*)d_in[3],  (const float*)d_in[9],
                        (const float*)d_in[15], (const float*)d_in[21]};
  const float* AD[4] = {(const float*)d_in[4],  (const float*)d_in[10],
                        (const float*)d_in[16], (const float*)d_in[22]};
  const float* BI[4] = {(const float*)d_in[5],  (const float*)d_in[11],
                        (const float*)d_in[17], (const float*)d_in[23]};
  const float* GA[3] = {(const float*)d_in[6],  (const float*)d_in[12],
                        (const float*)d_in[18]};
  const float* BE[3] = {(const float*)d_in[7],  (const float*)d_in[13],
                        (const float*)d_in[19]};

  // workspace bump allocator (256B aligned)
  char* base = (char*)d_ws;
  size_t off = 0;
  auto alloc = [&](size_t bytes) -> void* {
    void* p = base + off;
    off = (off + bytes + 255) & ~(size_t)255;
    return p;
  };
  float*    h_feat = (float*)   alloc((size_t)NNODE * 512 * sizeof(float));
  float*    agg    = (float*)   alloc((size_t)NNODE * 512 * sizeof(float));
  float*    xbuf   = (float*)   alloc((size_t)NNODE * 128 * sizeof(float));
  float*    ybuf   = (float*)   alloc((size_t)NNODE * 128 * sizeof(float));
  float*    a_s    = (float*)   alloc((size_t)NNODE * HEADS * sizeof(float));
  float*    a_d    = (float*)   alloc((size_t)NNODE * HEADS * sizeof(float));
  unsigned* emax   = (unsigned*)alloc((size_t)NNODE * HEADS * sizeof(unsigned));
  float*    denom  = (float*)   alloc((size_t)NNODE * HEADS * sizeof(float));
  float*    eedge  = (float*)   alloc((size_t)ETOT  * HEADS * sizeof(float));
  float*    bnsum  = (float*)   alloc(128 * sizeof(float));
  float*    bnsqs  = (float*)   alloc(128 * sizeof(float));
  unsigned* smax   = (unsigned*)alloc(sizeof(unsigned));
  float*    ssum   = (float*)   alloc(sizeof(float));

  const unsigned ORD_NEG_INF = 0x007FFFFFu; // f32_ord(-inf)
  const int out_cs[4] = {128, 64, 32, 1};
  const int ROWBLKS = (NNODE / 16 + 7) / 8;            // 157

  const float* Xin = x;
  for (int l = 0; l < 4; l++) {
    int out_c = out_cs[l];
    int hc = HEADS * out_c;
    int ntc = (hc + 15) / 16;
    int gblocks = ntc * ROWBLKS;

    // dense transform via WMMA (compile-time IN_C/HC per layer)
    switch (l) {
      case 0: gemm_wmma_kernel<4,   512><<<gblocks, 256, 0, stream>>>(Xin, W[0], h_feat); break;
      case 1: gemm_wmma_kernel<128, 256><<<gblocks, 256, 0, stream>>>(Xin, W[1], h_feat); break;
      case 2: gemm_wmma_kernel<64,  128><<<gblocks, 256, 0, stream>>>(Xin, W[2], h_feat); break;
      case 3: gemm_wmma_kernel<32,  4  ><<<gblocks, 256, 0, stream>>>(Xin, W[3], h_feat); break;
    }

    // attention logits per (node, head)
    switch (l) {
      case 0: alpha_kernel<128><<<grid1d((long)NNODE * HEADS), 256, 0, stream>>>(h_feat, AS[0], AD[0], a_s, a_d); break;
      case 1: alpha_kernel<64 ><<<grid1d((long)NNODE * HEADS), 256, 0, stream>>>(h_feat, AS[1], AD[1], a_s, a_d); break;
      case 2: alpha_kernel<32 ><<<grid1d((long)NNODE * HEADS), 256, 0, stream>>>(h_feat, AS[2], AD[2], a_s, a_d); break;
      case 3: alpha_kernel<1  ><<<grid1d((long)NNODE * HEADS), 256, 0, stream>>>(h_feat, AS[3], AD[3], a_s, a_d); break;
    }

    // init per-layer accumulators
    fill_u32_kernel<<<grid1d((long)NNODE * HEADS), 256, 0, stream>>>(emax, ORD_NEG_INF, NNODE * HEADS);
    fill_u32_kernel<<<grid1d((long)NNODE * HEADS), 256, 0, stream>>>((unsigned*)denom, 0u, NNODE * HEADS);
    fill_u32_kernel<<<grid1d((long)NNODE * hc), 256, 0, stream>>>((unsigned*)agg, 0u, NNODE * hc);
    fill_u32_kernel<<<grid1d(out_c), 256, 0, stream>>>((unsigned*)bnsum, 0u, out_c);
    fill_u32_kernel<<<grid1d(out_c), 256, 0, stream>>>((unsigned*)bnsqs, 0u, out_c);

    // segment softmax + scatter-aggregate
    edge_max_kernel<<<grid1d((long)ETOT * HEADS), 256, 0, stream>>>(ei, a_s, a_d, eedge, emax);
    edge_exp_kernel<<<grid1d((long)ETOT * HEADS), 256, 0, stream>>>(ei, eedge, emax, denom);
    switch (l) {
      case 0: edge_scatter_kernel<128><<<grid1d((long)ETOT * HEADS), 256, 0, stream>>>(ei, eedge, denom, h_feat, agg); break;
      case 1: edge_scatter_kernel<64 ><<<grid1d((long)ETOT * HEADS), 256, 0, stream>>>(ei, eedge, denom, h_feat, agg); break;
      case 2: edge_scatter_kernel<32 ><<<grid1d((long)ETOT * HEADS), 256, 0, stream>>>(ei, eedge, denom, h_feat, agg); break;
      case 3: edge_scatter_kernel<1  ><<<grid1d((long)ETOT * HEADS), 256, 0, stream>>>(ei, eedge, denom, h_feat, agg); break;
    }

    // head-mean + bias (+ BN stats for layers 0..2)
    node_finalize_kernel<<<grid1d((long)NNODE * out_c), 256, 0, stream>>>(
        agg, BI[l], ybuf, bnsum, bnsqs, out_c, hc, (l < 3) ? 1 : 0);

    if (l < 3) {
      bn_act_kernel<<<grid1d((long)NNODE * out_c), 256, 0, stream>>>(
          ybuf, bnsum, bnsqs, GA[l], BE[l], xbuf, out_c,
          (l < 2) ? 1 : 0, 0x9E3779B9u * (unsigned)(l + 1));
      Xin = xbuf;
    }
  }

  // layer-4 actor is in ybuf[n] (out_c == 1): tanh -> softmax over N
  float* out = (float*)d_out;
  fill_u32_kernel<<<1, 1, 0, stream>>>(smax, ORD_NEG_INF, 1);
  fill_u32_kernel<<<1, 1, 0, stream>>>((unsigned*)ssum, 0u, 1);
  tanh_max_kernel<<<grid1d(NNODE), 256, 0, stream>>>(ybuf, out, smax);
  sum_exp_kernel<<<grid1d(NNODE), 256, 0, stream>>>(out, smax, ssum);
  prob_kernel<<<grid1d(NNODE), 256, 0, stream>>>(out, smax, ssum);
}
